// DNCWrapper_69621419868302
// MI455X (gfx1250) — matrix operand order repaired and checked by reading, not compiled, not run on hardware
//
#include <hip/hip_runtime.h>
#include <cstdint>
#include <cstddef>

// ---------------- problem constants ----------------
#define Bsz   128
#define Tsz   32
#define INsz  512
#define Hsz   512
#define FH    2048          // 4*H
#define Msz   16
#define Wcs   20
#define Rsz   4
#define RWs   80            // R*Wc
#define NNs   592           // IN + RW
#define NNP   608           // padded to multiple of 32 for WMMA K
#define IFs   163           // interface vector length
#define IFP   176           // padded to multiple of 16 for WMMA N
#define Lsz   2
#define CLIPV 20.0f
#define EPSV  1e-6f
#define DELTAV 5e-6f

// ---------------- WMMA types ----------------
typedef __attribute__((ext_vector_type(16))) __bf16 v16bf;
typedef __attribute__((ext_vector_type(8)))  float  v8f;

union FragBF { uint32_t u[8]; v16bf v; };

__device__ __forceinline__ uint16_t f2bf(float f) {
  uint32_t u = __float_as_uint(f);
  uint32_t r = (u + 0x7fffu + ((u >> 16) & 1u)) >> 16;   // round-nearest-even
  return (uint16_t)r;
}
__device__ __forceinline__ float sigm(float x) { return 1.f / (1.f + expf(-x)); }
__device__ __forceinline__ float softplus(float x) {
  return fmaxf(x, 0.f) + log1pf(expf(-fabsf(x)));
}

// ---------------- weight fp32 -> bf16 (with row/K zero padding) ----------------
__global__ void cvt_bf16_pad(const float* __restrict__ src, uint16_t* __restrict__ dst,
                             int rIn, int rOut, int kIn, int kOut, int total) {
  int idx = blockIdx.x * 256 + threadIdx.x;
  if (idx >= total) return;
  int k  = idx % kOut;
  int rl = idx / kOut;
  int r  = rl % rOut;
  int l  = rl / rOut;
  float v = (r < rIn && k < kIn) ? src[((size_t)l * rIn + r) * kIn + k] : 0.f;
  dst[idx] = f2bf(v);
}

__global__ void pad_bias(const float* __restrict__ src, float* __restrict__ dst,
                         int nIn, int nOut, int total) {
  int idx = blockIdx.x * 256 + threadIdx.x;
  if (idx >= total) return;
  int n = idx % nOut;
  int l = idx / nOut;
  dst[idx] = (n < nIn) ? src[(size_t)l * nIn + n] : 0.f;
}

__global__ void zero_f32(float* __restrict__ p, int n) {
  int i = blockIdx.x * 256 + threadIdx.x;
  if (i < n) p[i] = 0.f;
}

// copy x[:, t, :] into bf16 layer-0 input buffer cols [0,512); cols 512..607 stay zero
__global__ void copy_x(const float* __restrict__ x, uint16_t* __restrict__ xbf, int t) {
  int idx = blockIdx.x * 256 + threadIdx.x;      // B*IN
  int b = idx >> 9, j = idx & 511;
  xbf[(size_t)b * NNP + j] = f2bf(x[((size_t)b * Tsz + t) * INsz + j]);
}

// ---------------- WMMA dual-source GEMM (bf16 activations & weights, fp32 accum) -----
// out[row, n] = sum_k A1[row,k]*W1[n,k] + sum_k A2[row,k]*W2[n,k] + b1[n] + b2[n]
// A bf16 row-major (lda row stride), W bf16 row-major [N][K].  Brows fixed = 128.
// grid.x = 8 (row tiles of 16), grid.y = ceil(N/(128*TP)).
// 8 waves/block; each wave owns TP consecutive 16x16 tiles sharing one A fragment.
// All TP B-fragments are loaded before the WMMA chain so the in-order load returns
// are covered by a single wait and the TP WMMAs issue back-to-back.
template <int TP>
__global__ __launch_bounds__(256) void gemm_wmma(
    const uint16_t* __restrict__ A1, int lda1, int K1, const uint16_t* __restrict__ W1,
    const uint16_t* __restrict__ A2, int lda2, int K2, const uint16_t* __restrict__ W2,
    const float* __restrict__ b1, const float* __restrict__ b2,
    float* __restrict__ out, int ldo, int Ncols) {
  const int wave = threadIdx.x >> 5;
  const int lane = threadIdx.x & 31;
  const int row0 = blockIdx.x * 16;
  const int ncol0 = blockIdx.y * (128 * TP) + wave * (16 * TP);
  if (ncol0 >= Ncols) return;                    // wave-uniform skip (EXEC stays full)

  const int hh = lane >> 4;                      // wave half
  const int l16 = lane & 15;

  v8f acc[TP];
#pragma unroll
  for (int tp = 0; tp < TP; ++tp) acc[tp] = {};

  for (int s = 0; s < 2; ++s) {
    const uint16_t* A = s ? A2 : A1;
    if (!A) continue;
    const int lda = s ? lda2 : lda1;
    const int K   = s ? K2 : K1;
    const uint16_t* W = s ? W2 : W1;
    const uint16_t* arow = A + (size_t)(row0 + l16) * lda;
    const uint16_t* wrow[TP];
#pragma unroll
    for (int tp = 0; tp < TP; ++tp)
      wrow[tp] = W + (size_t)(ncol0 + tp * 16 + l16) * K;

#pragma unroll 2
    for (int k = 0; k < K; k += 32) {
      // ---- A fragment (16x32 bf16): lane half hh needs K groups
      //      {hh*8 .. hh*8+7} and {16+hh*8 .. 16+hh*8+7}, each 16B contiguous.
      const uint16_t* ap = arow + k + hh * 8;
      uint4 a0 = *(const uint4*)(ap);
      uint4 a1 = *(const uint4*)(ap + 16);
      FragBF fa;
      fa.u[0] = a0.x; fa.u[1] = a0.y; fa.u[2] = a0.z; fa.u[3] = a0.w;
      fa.u[4] = a1.x; fa.u[5] = a1.y; fa.u[6] = a1.z; fa.u[7] = a1.w;

      // ---- issue ALL B-fragment loads first (32x16 bf16 each: lane n=l16,
      //      K groups {hh*8..+7} and {16+hh*8..+7}, same as A), then run the WMMA chain.
      FragBF fb[TP];
#pragma unroll
      for (int tp = 0; tp < TP; ++tp) {
        const uint16_t* wp = wrow[tp] + k + hh * 8;
        uint4 w0 = *(const uint4*)(wp);
        uint4 w1 = *(const uint4*)(wp + 16);
        __builtin_prefetch((const void*)(wp + 32), 0, 1);  // next K chunk
        fb[tp].u[0] = w0.x; fb[tp].u[1] = w0.y; fb[tp].u[2] = w0.z; fb[tp].u[3] = w0.w;
        fb[tp].u[4] = w1.x; fb[tp].u[5] = w1.y; fb[tp].u[6] = w1.z; fb[tp].u[7] = w1.w;
      }
#pragma unroll
      for (int tp = 0; tp < TP; ++tp)
        acc[tp] = __builtin_amdgcn_wmma_f32_16x16x32_bf16(
            /*neg_a=*/false, fa.v, /*neg_b=*/false, fb[tp].v,
            /*c_mod=*/(short)0, acc[tp], /*reuse_a=*/false, /*reuse_b=*/false);
    }
  }

  // C/D layout: VGPR r -> lanes 0-15: (M=r, N=lane); lanes 16-31: (M=8+r, N=lane-16)
  const int rbase = hh * 8;
#pragma unroll
  for (int tp = 0; tp < TP; ++tp) {
    const int col = ncol0 + tp * 16 + l16;
    float bias = (b1 ? b1[col] : 0.f) + (b2 ? b2[col] : 0.f);
#pragma unroll
    for (int r = 0; r < 8; ++r)
      out[(size_t)(row0 + rbase + r) * ldo + col] = acc[tp][r] + bias;
  }
}

// ---------------- LSTM gate nonlinearity ----------------
// g[b, 0:4H] = [i f g o]; c,h (fp32 masters) updated in place; bf16 mirror of h
// always written; optional clipped bf16 copy into concat buffer cols [0,512).
__global__ void lstm_gates(const float* __restrict__ g, float* __restrict__ c,
                           float* __restrict__ h, uint16_t* __restrict__ hbf,
                           uint16_t* __restrict__ outc) {
  int idx = blockIdx.x * 256 + threadIdx.x;      // B*H
  int b = idx >> 9, j = idx & 511;
  const float* gr = g + (size_t)b * FH;
  float gi = sigm(gr[j]);
  float gf = sigm(gr[Hsz + j]);
  float gg = tanhf(gr[2 * Hsz + j]);
  float go = sigm(gr[3 * Hsz + j]);
  float cn = gf * c[idx] + gi * gg;
  float hn = go * tanhf(cn);
  c[idx] = cn;
  h[idx] = hn;
  hbf[idx] = f2bf(hn);
  if (outc) outc[(size_t)b * NNP + j] = f2bf(fminf(fmaxf(hn, -CLIPV), CLIPV));
}

// ---------------- DNC memory step: one wave32 per batch element ----------------
__device__ __forceinline__ void softmax16(float* s, int t) {
  __syncthreads();
  float nv = 0.f;
  if (t < Msz) {
    float mx = -1e30f;
    for (int j = 0; j < Msz; ++j) mx = fmaxf(mx, s[j]);
    float den = 0.f;
    for (int j = 0; j < Msz; ++j) den += expf(s[j] - mx);
    nv = expf(s[t] - mx) / den;
  }
  __syncthreads();
  if (t < Msz) s[t] = nv;
  __syncthreads();
}

__global__ __launch_bounds__(32) void mem_step_kernel(
    const float* __restrict__ iface,
    float* __restrict__ memv, float* __restrict__ linkv, float* __restrict__ precv,
    float* __restrict__ rwv, float* __restrict__ wwv, float* __restrict__ usv,
    uint16_t* __restrict__ inpn) {
  const int b = blockIdx.x;
  const int t = threadIdx.x;
  const float* ifc = iface + (size_t)b * IFP;
  float* Mm = memv + (size_t)b * Msz * Wcs;
  float* Lk = linkv + (size_t)b * Msz * Msz;
  float* Pc = precv + (size_t)b * Msz;
  float* Rw = rwv  + (size_t)b * Rsz * Msz;
  float* Ww = wwv  + (size_t)b * Msz;
  float* Us = usv  + (size_t)b * Msz;

  __shared__ float s_rwold[Rsz][Msz];
  __shared__ float s_u[Msz], s_wcw[Msz], s_alloc[Msz], s_ww[Msz], s_prec[Msz], s_mninv[Msz];
  __shared__ float s_rcw[Rsz][Msz], s_rwnew[Rsz][Msz];

  if (t < Msz) {
    for (int r = 0; r < Rsz; ++r) s_rwold[r][t] = Rw[r * Msz + t];
    s_prec[t] = Pc[t];
  }
  __syncthreads();

  // usage retention + write-content score
  if (t < Msz) {
    float us = Us[t], w_old = Ww[t];
    us = us + (1.f - us) * w_old;
    float psi = 1.f;
    for (int r = 0; r < Rsz; ++r) psi *= 1.f - sigm(ifc[145 + r]) * s_rwold[r][t];
    us *= psi;
    Us[t] = us;
    s_u[t] = DELTAV + (1.f - DELTAV) * us;

    float nm2 = 0.f, nk2 = 0.f, dot = 0.f;
    for (int w = 0; w < Wcs; ++w) {
      float mv = Mm[t * Wcs + w];
      float kv = tanhf(ifc[84 + w]);
      nm2 += mv * mv; nk2 += kv * kv; dot += mv * kv;
    }
    float cs = dot / ((sqrtf(nm2) + EPSV) * (sqrtf(nk2) + EPSV));
    s_wcw[t] = cs * softplus(ifc[104]);
  }
  softmax16(&s_wcw[0], t);

  // allocation weights via stable rank (argsort of u, cumulative product)
  if (t < Msz) {
    float u = s_u[t];
    float pr = 1.f;
    for (int j = 0; j < Msz; ++j) {
      float uj = s_u[j];
      if (uj < u || (uj == u && j < t)) pr *= uj;
    }
    s_alloc[t] = (1.f - u) * pr;
    float ag = sigm(ifc[149]), wg = sigm(ifc[150]);
    float wn = wg * (ag * s_alloc[t] + (1.f - ag) * s_wcw[t]);
    s_ww[t] = wn;
    Ww[t] = wn;
  }
  __syncthreads();

  // memory write
  for (int idx = t; idx < Msz * Wcs; idx += 32) {
    int m = idx / Wcs, w = idx % Wcs;
    float e = sigm(ifc[105 + w]);
    float v = tanhf(ifc[125 + w]);
    float mval = Mm[idx];
    Mm[idx] = mval * (1.f - s_ww[m] * e) + s_ww[m] * v;
  }
  __syncthreads();

  // temporal link + precedence
  for (int idx = t; idx < Msz * Msz; idx += 32) {
    int i = idx / Msz, j = idx % Msz;
    float lk = Lk[idx];
    lk = (1.f - s_ww[i] - s_ww[j]) * lk + s_ww[i] * s_prec[j];
    Lk[idx] = (i == j) ? 0.f : lk;
  }
  if (t < Msz) {
    float sw = 0.f;
    for (int j = 0; j < Msz; ++j) sw += s_ww[j];
    Pc[t] = (1.f - sw) * s_prec[t] + s_ww[t];
  }
  __syncthreads();

  // new memory inverse norms
  if (t < Msz) {
    float nm2 = 0.f;
    for (int w = 0; w < Wcs; ++w) { float mv = Mm[t * Wcs + w]; nm2 += mv * mv; }
    s_mninv[t] = 1.f / (sqrtf(nm2) + EPSV);
  }
  __syncthreads();

  // read-content weights
  for (int r = 0; r < Rsz; ++r) {
    if (t < Msz) {
      float nk2 = 0.f, dot = 0.f;
      for (int w = 0; w < Wcs; ++w) {
        float kv = tanhf(ifc[r * Wcs + w]);
        nk2 += kv * kv; dot += Mm[t * Wcs + w] * kv;
      }
      float cs = dot * s_mninv[t] / (sqrtf(nk2) + EPSV);
      s_rcw[r][t] = cs * softplus(ifc[80 + r]);
    }
    softmax16(&s_rcw[r][0], t);
  }

  // forward/backward weights and read-mode mixing
  if (t < Msz) {
    for (int r = 0; r < Rsz; ++r) {
      float fw = 0.f, bw = 0.f;
      for (int j = 0; j < Msz; ++j) fw += Lk[t * Msz + j] * s_rwold[r][j];
      for (int i = 0; i < Msz; ++i) bw += s_rwold[r][i] * Lk[i * Msz + t];
      float e0 = ifc[151 + r * 3 + 0], e1 = ifc[151 + r * 3 + 1], e2 = ifc[151 + r * 3 + 2];
      float mx = fmaxf(e0, fmaxf(e1, e2));
      float x0 = expf(e0 - mx), x1 = expf(e1 - mx), x2 = expf(e2 - mx);
      float rn = (x0 * bw + x1 * fw + x2 * s_rcw[r][t]) / (x0 + x1 + x2);
      s_rwnew[r][t] = rn;
      Rw[r * Msz + t] = rn;
    }
  }
  __syncthreads();

  // read vectors (bf16) -> next-layer concat buffer cols [512, 592)
  for (int idx = t; idx < RWs; idx += 32) {
    int r = idx / Wcs, w = idx % Wcs;
    float v = 0.f;
    for (int m = 0; m < Msz; ++m) v += s_rwnew[r][m] * Mm[m * Wcs + w];
    inpn[(size_t)b * NNP + Hsz + idx] = f2bf(v);
  }
}

// ---------------- host-side orchestration ----------------
extern "C" void kernel_launch(void* const* d_in, const int* in_sizes, int n_in,
                              void* d_out, int out_size, void* d_ws, size_t ws_size,
                              hipStream_t stream) {
  (void)in_sizes; (void)n_in; (void)out_size; (void)ws_size;
  const float* x    = (const float*)d_in[0];
  const float* Wih0 = (const float*)d_in[1];
  const float* Whh0 = (const float*)d_in[2];
  const float* bih0 = (const float*)d_in[3];
  const float* bhh0 = (const float*)d_in[4];
  const float* Wih1 = (const float*)d_in[5];
  const float* Whh1 = (const float*)d_in[6];
  const float* bih1 = (const float*)d_in[7];
  const float* bhh1 = (const float*)d_in[8];
  const float* Wif  = (const float*)d_in[9];
  const float* bif  = (const float*)d_in[10];
  const float* Wout = (const float*)d_in[11];
  const float* bout = (const float*)d_in[12];
  float* y = (float*)d_out;

  // ---- workspace layout (256B aligned; zero-region buffers are 256B-multiple sized) ----
  char* p = (char*)d_ws;
  auto take = [&](size_t bytes) -> char* {
    char* r = p;
    p += (bytes + 255) & ~(size_t)255;
    return r;
  };
  uint16_t* Wih0b = (uint16_t*)take((size_t)Lsz * FH * NNP * 2);
  uint16_t* Whh0b = (uint16_t*)take((size_t)Lsz * FH * Hsz * 2);
  uint16_t* Wih1b = (uint16_t*)take((size_t)Lsz * FH * Hsz * 2);
  uint16_t* Whh1b = (uint16_t*)take((size_t)Lsz * FH * Hsz * 2);
  uint16_t* Wifb  = (uint16_t*)take((size_t)Lsz * IFP * Hsz * 2);
  uint16_t* Woutb = (uint16_t*)take((size_t)INsz * NNP * 2);

  // ---- contiguous zero-initialized region (state masters + bf16 activation buffers)
  float*    hbuf  = (float*)take((size_t)Lsz * 2 * Bsz * Hsz * 4);
  float*    cbuf  = (float*)take((size_t)Lsz * 2 * Bsz * Hsz * 4);
  uint16_t* hbfb  = (uint16_t*)take((size_t)Lsz * 2 * Bsz * Hsz * 2);
  float*    memb  = (float*)take((size_t)Lsz * Bsz * Msz * Wcs * 4);
  float*    linkb = (float*)take((size_t)Lsz * Bsz * Msz * Msz * 4);
  float*    precb = (float*)take((size_t)Lsz * Bsz * Msz * 4);
  float*    rwb   = (float*)take((size_t)Lsz * Bsz * Rsz * Msz * 4);
  float*    wwb   = (float*)take((size_t)Lsz * Bsz * Msz * 4);
  float*    usgb  = (float*)take((size_t)Lsz * Bsz * Msz * 4);
  uint16_t* xbf   = (uint16_t*)take((size_t)Bsz * NNP * 2);
  uint16_t* c1bf  = (uint16_t*)take((size_t)Bsz * NNP * 2);
  uint16_t* c2bf  = (uint16_t*)take((size_t)Bsz * NNP * 2);
  const int zero_elems = (int)(((char*)p - (char*)hbuf) / 4);

  float* gbuf  = (float*)take((size_t)Bsz * FH * 4);
  float* ifbuf = (float*)take((size_t)Bsz * IFP * 4);
  float* bifp  = (float*)take((size_t)Lsz * IFP * 4);

  // ---- per-call init (deterministic) ----
  zero_f32<<<(zero_elems + 255) / 256, 256, 0, stream>>>(hbuf, zero_elems);

  {
    int tot;
    tot = Lsz * FH * NNP;
    cvt_bf16_pad<<<(tot + 255) / 256, 256, 0, stream>>>(Wih0, Wih0b, FH, FH, NNs, NNP, tot);
    tot = Lsz * FH * Hsz;
    cvt_bf16_pad<<<(tot + 255) / 256, 256, 0, stream>>>(Whh0, Whh0b, FH, FH, Hsz, Hsz, tot);
    cvt_bf16_pad<<<(tot + 255) / 256, 256, 0, stream>>>(Wih1, Wih1b, FH, FH, Hsz, Hsz, tot);
    cvt_bf16_pad<<<(tot + 255) / 256, 256, 0, stream>>>(Whh1, Whh1b, FH, FH, Hsz, Hsz, tot);
    tot = Lsz * IFP * Hsz;
    cvt_bf16_pad<<<(tot + 255) / 256, 256, 0, stream>>>(Wif, Wifb, IFs, IFP, Hsz, Hsz, tot);
    tot = INsz * NNP;
    cvt_bf16_pad<<<(tot + 255) / 256, 256, 0, stream>>>(Wout, Woutb, INsz, INsz, NNs, NNP, tot);
    tot = Lsz * IFP;
    pad_bias<<<(tot + 255) / 256, 256, 0, stream>>>(bif, bifp, IFs, IFP, tot);
  }

  const dim3 blk(256);
  const dim3 gridG(8, FH / 256);     // TP=2: 8 waves x 32 cols = 256 cols/block
  const dim3 gridI(8, (IFP + 127) / 128);   // TP=1
  const dim3 gridO(8, INsz / 256);   // TP=2
  const int elemBH = Bsz * Hsz;      // 65536 -> 256 blocks

  for (int t = 0; t < Tsz; ++t) {
    copy_x<<<(Bsz * INsz) / 256, 256, 0, stream>>>(x, xbf, t);
    const uint16_t* bufin = xbf;
    for (int l = 0; l < Lsz; ++l) {
      uint16_t* bufout = (l == 0) ? c1bf : c2bf;
      float*    h0  = hbuf + (size_t)(l * 2 + 0) * Bsz * Hsz;
      float*    h1  = hbuf + (size_t)(l * 2 + 1) * Bsz * Hsz;
      float*    c0  = cbuf + (size_t)(l * 2 + 0) * Bsz * Hsz;
      float*    c1  = cbuf + (size_t)(l * 2 + 1) * Bsz * Hsz;
      uint16_t* h0b = hbfb + (size_t)(l * 2 + 0) * Bsz * Hsz;
      uint16_t* h1b = hbfb + (size_t)(l * 2 + 1) * Bsz * Hsz;

      // cell 0: g = inp @ Wih0^T + h0 @ Whh0^T + b_ih0 + b_hh0
      gemm_wmma<2><<<gridG, blk, 0, stream>>>(
          bufin, NNP, NNP, Wih0b + (size_t)l * FH * NNP,
          h0b, Hsz, Hsz, Whh0b + (size_t)l * FH * Hsz,
          bih0 + (size_t)l * FH, bhh0 + (size_t)l * FH,
          gbuf, FH, FH);
      lstm_gates<<<elemBH / 256, 256, 0, stream>>>(gbuf, c0, h0, h0b, nullptr);

      // cell 1: g = h0 @ Wih1^T + h1 @ Whh1^T + b_ih1 + b_hh1
      gemm_wmma<2><<<gridG, blk, 0, stream>>>(
          h0b, Hsz, Hsz, Wih1b + (size_t)l * FH * Hsz,
          h1b, Hsz, Hsz, Whh1b + (size_t)l * FH * Hsz,
          bih1 + (size_t)l * FH, bhh1 + (size_t)l * FH,
          gbuf, FH, FH);
      lstm_gates<<<elemBH / 256, 256, 0, stream>>>(gbuf, c1, h1, h1b, bufout);

      // interface vector: iface = out @ W_iface^T + b_iface (padded to 176 cols)
      gemm_wmma<1><<<gridI, blk, 0, stream>>>(
          bufout, NNP, Hsz, Wifb + (size_t)l * IFP * Hsz,
          nullptr, 0, 0, nullptr,
          bifp + (size_t)l * IFP, nullptr,
          ifbuf, IFP, IFP);

      // DNC memory update; writes bf16 read vectors into bufout cols [512,592)
      mem_step_kernel<<<Bsz, 32, 0, stream>>>(
          ifbuf,
          memb  + (size_t)l * Bsz * Msz * Wcs,
          linkb + (size_t)l * Bsz * Msz * Msz,
          precb + (size_t)l * Bsz * Msz,
          rwb   + (size_t)l * Bsz * Rsz * Msz,
          wwb   + (size_t)l * Bsz * Msz,
          usgb  + (size_t)l * Bsz * Msz,
          bufout);

      bufin = bufout;
    }
    // y_t = concat(out, read) @ W_out^T + b_out  -> d_out[:, t, :]
    gemm_wmma<2><<<gridO, blk, 0, stream>>>(
        c2bf, NNP, NNP, Woutb,
        nullptr, 0, 0, nullptr,
        bout, nullptr,
        y + (size_t)t * INsz, Tsz * INsz, INsz);
  }
}
